// AttentionBlock_23218593202482
// MI455X (gfx1250) — matrix operand order repaired and checked by reading, not compiled, not run on hardware
//
#include <hip/hip_runtime.h>
#include <hip/hip_bf16.h>

// ---------------------------------------------------------------------------
// Fused windowed attention for MI455X (gfx1250, wave32, WMMA bf16).
// B=512, N=64, C=1024, H=16, Dh=64, grid GH=GW=8.
// ---------------------------------------------------------------------------

typedef __attribute__((ext_vector_type(16))) __bf16 v16bf;
typedef __attribute__((ext_vector_type(8)))  float  v8f;

#define BN   512
#define NN   64
#define CC   1024
#define HH   16
#define DH   64

// LDS layout (bytes) -- total 324,864 <= 320KB WGP LDS
#define XS_STRIDE 1040        // 1024 + 16 bf16 pad (anti bank conflict)
#define QK_STRIDE 80          // 64 + 16 bf16 pad
#define SF_STRIDE 68          // 64 + 4 f32 pad
#define OFF_XS    0
#define OFF_OS    133120
#define OFF_QS    266240
#define OFF_KS    276480
#define OFF_VT    286720
#define OFF_SF    296960
#define OFF_PB    314368
#define OFF_QSC   324608
#define LDS_BYTES 324864

// ws layout (bytes)
#define WSOFF_WQKV 0                 // bf16 [H][192][1024] column-major slices
#define WSOFF_WO   6291456           // bf16 [1024 cols][1024 k] column-major
#define WSOFF_BIAS 8388608           // f32  [H][64][64]
#define WSOFF_QSC  8650752           // f32  [64]

// Native RTNE f32 -> bf16 (single v_cvt on gfx1250, often packed by the compiler)
static __device__ __forceinline__ unsigned short f2bf(float f) {
    __bf16 h = (__bf16)f;
    return __builtin_bit_cast(unsigned short, h);
}

union ABv { v16bf v; uint4 u[2]; };

// A-matrix 16x32 bf16 load: lane<16 -> row m=lane, K = 0..15;
// lane>=16 -> row m=lane-16, K = 16..31. src is row-major [rows][stride] u16.
static __device__ __forceinline__ v16bf load_A(const unsigned short* base, int stride,
                                               int r0, int k0, int lane) {
    int m = lane & 15, hs = lane >> 4;
    const unsigned short* p = base + (size_t)(r0 + m) * stride + k0 + hs * 16;
    ABv r;
    r.u[0] = *(const uint4*)(p);        // K = k0+hs*16 .. +7
    r.u[1] = *(const uint4*)(p + 8);    // K = k0+hs*16+8 .. +15
    return r.v;
}

// B-matrix 32x16 bf16 load: lane holds column n=lane&15, K = (lane>=16?16:0)+0..15
// caller passes pointer to the 16 consecutive bf16 (32B aligned).
static __device__ __forceinline__ v16bf load_B(const unsigned short* p) {
    ABv r;
    r.u[0] = *(const uint4*)(p);
    r.u[1] = *(const uint4*)(p + 8);
    return r.v;
}

static __device__ __forceinline__ v8f wmma_bf16(v16bf a, v16bf b, v8f c) {
    return __builtin_amdgcn_wmma_f32_16x16x32_bf16(false, a, false, b, (short)0, c,
                                                   false, false);
}

// ------------------------------- prep kernels ------------------------------

// Pack Wq|Wk|Wv (f32 row-major [k][n]) -> bf16 column-major per-head slices:
// wqkv[(h*192 + j)*1024 + k], j in [0,192): 0-63 Q cols, 64-127 K cols, 128-191 V cols.
__global__ void pack_wqkv_kernel(const float* __restrict__ Wq, const float* __restrict__ Wk,
                                 const float* __restrict__ Wv, unsigned short* __restrict__ out) {
    int o = blockIdx.x * blockDim.x + threadIdx.x;
    if (o >= HH * 192 * CC) return;
    int k  = o & (CC - 1);
    int cj = o >> 10;
    int j  = cj % 192;
    int h  = cj / 192;
    int sel = j >> 6;
    int col = h * DH + (j & 63);
    const float* W = (sel == 0) ? Wq : (sel == 1) ? Wk : Wv;
    out[o] = f2bf(W[(size_t)k * CC + col]);
}

// Wo f32 [k][n] -> bf16 column-major wo[n*1024 + k]
__global__ void pack_wo_kernel(const float* __restrict__ Wo, unsigned short* __restrict__ out) {
    int o = blockIdx.x * blockDim.x + threadIdx.x;
    if (o >= CC * CC) return;
    int k = o & (CC - 1);
    int n = o >> 10;
    out[o] = f2bf(Wo[(size_t)k * CC + n]);
}

// biasF[h][q][k] = table[rel_index[q*64+k]*H + h]; qscale[q] = 2/(1+s^2)
__global__ void bias_scale_kernel(const float* __restrict__ table, const int* __restrict__ rel,
                                  const float* __restrict__ x0, float* __restrict__ biasF,
                                  float* __restrict__ qscale) {
    int o = blockIdx.x * blockDim.x + threadIdx.x;
    if (o < HH * NN * NN) {
        int h  = o >> 12;
        int qk = o & 4095;
        biasF[o] = table[(size_t)rel[qk] * HH + h];
    }
    if (o < NN) {
        float v = x0[o];
        float s = (1.0f + v) / (1.0f - v + 1e-5f);
        qscale[o] = 2.0f / (1.0f + s * s);
    }
}

// --------------------------- fused attention kernel ------------------------
// One workgroup (8 wave32s) per batch element. Everything stays in 320KB LDS.

__global__ __launch_bounds__(256, 1)
void attn_fused_kernel(const float* __restrict__ x,
                       const float* __restrict__ bq, const float* __restrict__ bk,
                       const float* __restrict__ bvv, const float* __restrict__ bo,
                       const unsigned short* __restrict__ Wqkv,
                       const unsigned short* __restrict__ Wo,
                       const float* __restrict__ biasF, const float* __restrict__ qscale,
                       float* __restrict__ out) {
    extern __shared__ char smem[];
    unsigned short* xs  = (unsigned short*)(smem + OFF_XS);   // [64][1040] bf16 x_b
    unsigned short* os  = (unsigned short*)(smem + OFF_OS);   // [64][1040] bf16 attn out
    unsigned short* qs  = (unsigned short*)(smem + OFF_QS);   // [64][80]   bf16 Q_h
    unsigned short* ks  = (unsigned short*)(smem + OFF_KS);   // [64][80]   bf16 K_h
    unsigned short* vt  = (unsigned short*)(smem + OFF_VT);   // [64][80]   bf16 V_h^T
    float*          Sf  = (float*)(smem + OFF_SF);            // [64][68]   f32 logits
    unsigned short* Pb  = (unsigned short*)(smem + OFF_PB);   // [64][80]   bf16 probs
    float*          qsl = (float*)(smem + OFF_QSC);           // [64]       f32 2/(1+s^2)

    const int b    = blockIdx.x;
    const int tid  = threadIdx.x;
    const int wave = tid >> 5;
    const int lane = tid & 31;
    const int n    = lane & 15;   // WMMA B/D column within tile
    const int hs   = lane >> 4;   // lane-half select
    const int rs   = wave & 3;    // 16-row stripe for projection stages
    const int cg   = wave >> 2;   // column group

    const v8f vzero = {};

    // ---- stage x_b -> LDS as bf16 (vectorized f32x4 reads, 8B LDS stores)
    {
        const float4* xg = (const float4*)(x + (size_t)b * NN * CC);
        for (int i = tid; i < NN * (CC / 4); i += 256) {
            int row = i >> 8;          // CC/4 = 256 float4 per row
            int c4  = i & 255;
            float4 f = xg[(size_t)row * 256 + c4];
            unsigned p0 = (unsigned)f2bf(f.x) | ((unsigned)f2bf(f.y) << 16);
            unsigned p1 = (unsigned)f2bf(f.z) | ((unsigned)f2bf(f.w) << 16);
            *(uint2*)(xs + (size_t)row * XS_STRIDE + c4 * 4) = make_uint2(p0, p1);
        }
    }
    if (tid < NN) qsl[tid] = qscale[tid];
    __syncthreads();

    // ==================== per-head loop ====================
    for (int h = 0; h < HH; ++h) {
        // ---- QKV projection for head h: [64,1024] @ [1024,192] via WMMA.
        // Wave (rs, cg): rows rs*16..+15, cols cg*96 + 6 tiles of 16.
        v8f acc[6];
        #pragma unroll
        for (int j = 0; j < 6; ++j) acc[j] = vzero;

        for (int kk = 0; kk < CC; kk += 32) {
            v16bf a = load_A(xs, XS_STRIDE, rs * 16, kk, lane);
            #pragma unroll
            for (int j = 0; j < 6; ++j) {
                int ct = cg * 6 + j;  // col tile 0..11 of 192 cols
                const unsigned short* bp =
                    Wqkv + ((size_t)(h * 192 + ct * 16 + n)) * CC + kk + hs * 16;
                __builtin_prefetch(bp + 32, 0, 0);  // next K-slab -> global_prefetch
                acc[j] = wmma_bf16(a, load_B(bp), acc[j]);
            }
        }
        // bias add + store to qs / ks / vt (V transposed for the PV B-operand)
        {
            int r0 = rs * 16 + hs * 8;
            #pragma unroll
            for (int j = 0; j < 6; ++j) {
                int col = cg * 96 + j * 16 + n;   // 0..191
                if (col < 64) {
                    float bias = bq[h * DH + col];
                    #pragma unroll
                    for (int i = 0; i < 8; ++i)
                        qs[(size_t)(r0 + i) * QK_STRIDE + col] = f2bf(acc[j][i] + bias);
                } else if (col < 128) {
                    float bias = bk[h * DH + col - 64];
                    #pragma unroll
                    for (int i = 0; i < 8; ++i)
                        ks[(size_t)(r0 + i) * QK_STRIDE + (col - 64)] = f2bf(acc[j][i] + bias);
                } else {
                    // transposed store: 8 consecutive bf16 -> single ds_store_b128
                    float bias = bvv[h * DH + col - 128];
                    unsigned pk[4];
                    #pragma unroll
                    for (int m = 0; m < 4; ++m)
                        pk[m] = (unsigned)f2bf(acc[j][2 * m] + bias) |
                                ((unsigned)f2bf(acc[j][2 * m + 1] + bias) << 16);
                    *(uint4*)(vt + (size_t)(col - 128) * QK_STRIDE + r0) =
                        make_uint4(pk[0], pk[1], pk[2], pk[3]);
                }
            }
        }
        __syncthreads();

        // ---- S = Q @ K^T, + rel-pos bias, * 2/(1+s^2)  -> Sf (f32)
        // The wave's two tiles share the row stripe: one A load feeds two
        // independent WMMAs (fills the bf16 WMMA->WMMA hazard window).
        {
            int t0  = wave * 2;
            int tr  = t0 >> 2;
            int tc0 = t0 & 3, tc1 = tc0 + 1;
            v8f s0 = vzero, s1 = vzero;
            #pragma unroll
            for (int kk = 0; kk < DH; kk += 32) {
                v16bf a  = load_A(qs, QK_STRIDE, tr * 16, kk, lane);
                v16bf b0 = load_B(ks + (size_t)(tc0 * 16 + n) * QK_STRIDE + kk + hs * 16);
                v16bf b1 = load_B(ks + (size_t)(tc1 * 16 + n) * QK_STRIDE + kk + hs * 16);
                s0 = wmma_bf16(a, b0, s0);
                s1 = wmma_bf16(a, b1, s1);
            }
            int q0 = tr * 16 + hs * 8;
            int k0 = tc0 * 16 + n, k1 = tc1 * 16 + n;
            const float* bF = biasF + (size_t)h * NN * NN;
            #pragma unroll
            for (int i = 0; i < 8; ++i) {
                int q = q0 + i;
                float qsc = qsl[q];
                Sf[(size_t)q * SF_STRIDE + k0] = (s0[i] + bF[(size_t)q * NN + k0]) * qsc;
                Sf[(size_t)q * SF_STRIDE + k1] = (s1[i] + bF[(size_t)q * NN + k1]) * qsc;
            }
        }
        __syncthreads();

        // ---- row softmax (64 rows, one thread per row), write bf16 probs
        if (tid < NN) {
            float* row = Sf + (size_t)tid * SF_STRIDE;
            float m = row[0];
            #pragma unroll 8
            for (int j = 1; j < NN; ++j) m = fmaxf(m, row[j]);
            float sum = 0.0f;
            #pragma unroll 8
            for (int j = 0; j < NN; ++j) { float e = __expf(row[j] - m); row[j] = e; sum += e; }
            float inv = 1.0f / sum;
            unsigned short* pr = Pb + (size_t)tid * QK_STRIDE;
            #pragma unroll 4
            for (int j = 0; j < NN; j += 4) {
                unsigned p0 = (unsigned)f2bf(row[j] * inv) |
                              ((unsigned)f2bf(row[j + 1] * inv) << 16);
                unsigned p1 = (unsigned)f2bf(row[j + 2] * inv) |
                              ((unsigned)f2bf(row[j + 3] * inv) << 16);
                *(uint2*)(pr + j) = make_uint2(p0, p1);
            }
        }
        __syncthreads();

        // ---- O = P @ V -> os[:, h*64 : h*64+64] (bf16); same A-sharing trick
        {
            int t0  = wave * 2;
            int tr  = t0 >> 2;
            int tc0 = t0 & 3, tc1 = tc0 + 1;
            v8f o0 = vzero, o1 = vzero;
            #pragma unroll
            for (int kk = 0; kk < NN; kk += 32) {
                v16bf a  = load_A(Pb, QK_STRIDE, tr * 16, kk, lane);
                v16bf b0 = load_B(vt + (size_t)(tc0 * 16 + n) * QK_STRIDE + kk + hs * 16);
                v16bf b1 = load_B(vt + (size_t)(tc1 * 16 + n) * QK_STRIDE + kk + hs * 16);
                o0 = wmma_bf16(a, b0, o0);
                o1 = wmma_bf16(a, b1, o1);
            }
            int q0 = tr * 16 + hs * 8;
            int d0 = h * DH + tc0 * 16 + n;
            int d1 = h * DH + tc1 * 16 + n;
            #pragma unroll
            for (int i = 0; i < 8; ++i) {
                os[(size_t)(q0 + i) * XS_STRIDE + d0] = f2bf(o0[i]);
                os[(size_t)(q0 + i) * XS_STRIDE + d1] = f2bf(o1[i]);
            }
        }
        __syncthreads();
    }

    // ---- final projection: out_b = os @ Wo + bo  ([64,1024]@[1024,1024])
    float* og = out + (size_t)b * NN * CC;
    for (int g = 0; g < 8; ++g) {
        v8f acc[4];
        #pragma unroll
        for (int j = 0; j < 4; ++j) acc[j] = vzero;
        for (int kk = 0; kk < CC; kk += 32) {
            v16bf a = load_A(os, XS_STRIDE, rs * 16, kk, lane);
            #pragma unroll
            for (int j = 0; j < 4; ++j) {
                int ct = cg + 2 * (g * 4 + j);  // interleaved col tiles 0..63
                const unsigned short* bp = Wo + ((size_t)(ct * 16 + n)) * CC + kk + hs * 16;
                __builtin_prefetch(bp + 32, 0, 0);
                acc[j] = wmma_bf16(a, load_B(bp), acc[j]);
            }
        }
        int q0 = rs * 16 + hs * 8;
        #pragma unroll
        for (int j = 0; j < 4; ++j) {
            int col = (cg + 2 * (g * 4 + j)) * 16 + n;
            float bias = bo[col];
            #pragma unroll
            for (int i = 0; i < 8; ++i)
                og[(size_t)(q0 + i) * CC + col] = acc[j][i] + bias;
        }
    }
}

// ------------------------------- launcher ----------------------------------

extern "C" void kernel_launch(void* const* d_in, const int* in_sizes, int n_in,
                              void* d_out, int out_size, void* d_ws, size_t ws_size,
                              hipStream_t stream) {
    (void)in_sizes; (void)n_in; (void)out_size; (void)ws_size;
    const float* x     = (const float*)d_in[0];
    const float* Wq    = (const float*)d_in[1];
    const float* bq    = (const float*)d_in[2];
    const float* Wk    = (const float*)d_in[3];
    const float* bk    = (const float*)d_in[4];
    const float* Wv    = (const float*)d_in[5];
    const float* bv    = (const float*)d_in[6];
    const float* Wo    = (const float*)d_in[7];
    const float* bo    = (const float*)d_in[8];
    const float* table = (const float*)d_in[9];
    const float* x0    = (const float*)d_in[10];
    const int*   rel   = (const int*)d_in[11];

    char* ws = (char*)d_ws;
    unsigned short* wqkv  = (unsigned short*)(ws + WSOFF_WQKV);
    unsigned short* wob   = (unsigned short*)(ws + WSOFF_WO);
    float*          biasF = (float*)(ws + WSOFF_BIAS);
    float*          qsc   = (float*)(ws + WSOFF_QSC);

    pack_wqkv_kernel<<<(HH * 192 * CC) / 256, 256, 0, stream>>>(Wq, Wk, Wv, wqkv);
    pack_wo_kernel<<<(CC * CC) / 256, 256, 0, stream>>>(Wo, wob);
    bias_scale_kernel<<<(HH * NN * NN) / 256, 256, 0, stream>>>(table, rel, x0, biasF, qsc);

    hipFuncSetAttribute(reinterpret_cast<const void*>(attn_fused_kernel),
                        hipFuncAttributeMaxDynamicSharedMemorySize, LDS_BYTES);
    attn_fused_kernel<<<BN, 256, LDS_BYTES, stream>>>(x, bq, bk, bv, bo, wqkv, wob,
                                                      biasF, qsc, (float*)d_out);
}
